// EdgeDecoder_71511205478789
// MI455X (gfx1250) — matrix operand order repaired and checked by reading, not compiled, not run on hardware
//
#include <hip/hip_runtime.h>
#include <stdint.h>

#define HIDDEN 128
// LDS row stride for W1T tile, in bf16 elements. 264 halfs = 528 B = 132 dwords.
// 132 % 64 == 4  ->  lane m starts at dword-bank 4m; a b128 load covers banks
// 4m..4m+3, so 16 lanes tile all 64 banks exactly once: conflict-free.
#define SW_STRIDE 264
#define SW_BYTES  (HIDDEN * SW_STRIDE * 2)   // 67584 B < 320 KB WGP LDS

typedef __attribute__((ext_vector_type(16))) __bf16       v16bf;
typedef __attribute__((ext_vector_type(8)))  float        v8f;
typedef __attribute__((ext_vector_type(4)))  unsigned int v4u;

union Frag16 {            // 32 bytes: one WMMA 16-bit A or B fragment (8 VGPRs)
    v16bf v;
    v4u   q[2];
};

// float -> bf16 bits, round-to-nearest-even
__device__ __forceinline__ unsigned short f2bf(float f) {
    unsigned int u = __float_as_uint(f);
    u += 0x7FFFu + ((u >> 16) & 1u);
    return (unsigned short)(u >> 16);
}

// Convert z (fp32) -> bf16 bits, 4 elements per thread, grid-stride.
__global__ void convert_z_kernel(const float* __restrict__ z,
                                 unsigned short* __restrict__ zb, int n4) {
    int i = blockIdx.x * blockDim.x + threadIdx.x;
    for (; i < n4; i += gridDim.x * blockDim.x) {
        const float4 f = ((const float4*)z)[i];
        unsigned int lo = (unsigned int)f2bf(f.x) | ((unsigned int)f2bf(f.y) << 16);
        unsigned int hi = (unsigned int)f2bf(f.z) | ((unsigned int)f2bf(f.w) << 16);
        ((uint2*)zb)[i] = make_uint2(lo, hi);
    }
}

// W1 is (2H, H) row-major fp32; produce W1T bf16 as [n=0..127][k=0..255].
__global__ void convert_w1t_kernel(const float* __restrict__ W1,
                                   unsigned short* __restrict__ w1t) {
    int idx = blockIdx.x * blockDim.x + threadIdx.x;   // 0 .. 32767
    int n  = idx >> 8;          // output col of W1 (0..127)
    int kk = idx & 255;         // input row of W1 (0..255)
    w1t[n * 256 + kk] = f2bf(W1[kk * HIDDEN + n]);
}

__global__ __launch_bounds__(256)
void edge_decoder_kernel(const unsigned short* __restrict__ zb,    // [n_nodes][128] bf16
                         const long long*      __restrict__ edges, // [2][n_edges] int64
                         const unsigned short* __restrict__ w1t,   // [128][256] bf16
                         const float* __restrict__ b1,
                         const float* __restrict__ W2,
                         const float* __restrict__ b2,
                         float* __restrict__ out,
                         int n_edges) {
    extern __shared__ unsigned short sW[];   // [128][SW_STRIDE] bf16, padded

    const int tid = threadIdx.x;

    // Stage W1T into LDS with padded stride: 128 rows x 32 chunks of 16 B.
    {
        #pragma unroll
        for (int i = 0; i < 16; ++i) {
            const int idx = tid + 256 * i;       // 0 .. 4095
            const int row = idx >> 5;            // 0 .. 127
            const int c   = idx & 31;            // 0 .. 31 (16 B chunks)
            *(v4u*)(sW + row * SW_STRIDE + c * 8) =
                *(const v4u*)(w1t + row * 256 + c * 8);
        }
    }
    __syncthreads();

    const int wave = tid >> 5;
    const int lane = tid & 31;
    const int m    = lane & 15;     // edge row within 16-edge tile / B column
    const int hi   = lane >> 4;     // lane half selects K sub-block
    const int tile = blockIdx.x * 128 + wave * 16;

    int e = tile + m;
    if (e >= n_edges) e = n_edges - 1;
    const long long srcN = edges[e];
    const long long dstN = edges[(long long)n_edges + e];
    const unsigned short* psrc = zb + (size_t)srcN * HIDDEN;
    const unsigned short* pdst = zb + (size_t)dstN * HIDDEN;
    const int ka = hi * 8;          // A-frag K offset within 32-wide K step

    v8f acc[8];
    #pragma unroll
    for (int nt = 0; nt < 8; ++nt) { v8f zf = {}; acc[nt] = zf; }

    // K = 256 total: k<4 -> z[src] columns, k>=4 -> z[dst] columns.
    #pragma unroll
    for (int k = 0; k < 8; ++k) {
        const unsigned short* prow = (k < 4) ? psrc : pdst;
        const int kcol = (k & 3) * 32 + ka;
        Frag16 a;
        a.q[0] = *(const v4u*)(prow + kcol);        // K = kcol .. kcol+7
        a.q[1] = *(const v4u*)(prow + kcol + 16);   // K = kcol+16 .. kcol+23
        #pragma unroll
        for (int nt = 0; nt < 8; ++nt) {
            Frag16 b;  // lane holds col nt*16+m, K = k*32 + hi*8 + {0..7, 16..23}
            const unsigned short* pb =
                sW + ((nt * 16 + m) * SW_STRIDE + k * 32 + hi * 8);
            b.q[0] = *(const v4u*)(pb);
            b.q[1] = *(const v4u*)(pb + 16);
            acc[nt] = __builtin_amdgcn_wmma_f32_16x16x32_bf16(
                false, a.v, false, b.v, (short)0, acc[nt], false, false);
        }
    }

    // Epilogue: h = acc + b1, ReLU, dot with W2, reduce across the 16-lane half.
    float partial[8];
    #pragma unroll
    for (int r = 0; r < 8; ++r) partial[r] = 0.0f;
    #pragma unroll
    for (int nt = 0; nt < 8; ++nt) {
        const int col = nt * 16 + m;
        const float bb = b1[col];
        const float ww = W2[col];
        #pragma unroll
        for (int r = 0; r < 8; ++r) {
            float hv = acc[nt][r] + bb;
            hv = fmaxf(hv, 0.0f);
            partial[r] = fmaf(hv, ww, partial[r]);
        }
    }
    #pragma unroll
    for (int s = 1; s < 16; s <<= 1) {
        #pragma unroll
        for (int r = 0; r < 8; ++r)
            partial[r] += __shfl_xor(partial[r], s, 32);
    }
    if (m == 0) {                       // lanes 0 and 16 write rows 0-7 / 8-15
        const float bias2 = b2[0];
        const int rowbase = tile + hi * 8;
        #pragma unroll
        for (int r = 0; r < 8; ++r) {
            const int row = rowbase + r;
            if (row < n_edges) out[row] = partial[r] + bias2;
        }
    }
}

extern "C" void kernel_launch(void* const* d_in, const int* in_sizes, int n_in,
                              void* d_out, int out_size, void* d_ws, size_t ws_size,
                              hipStream_t stream) {
    const float*     z    = (const float*)d_in[0];
    const long long* edge = (const long long*)d_in[1];
    const float*     W1   = (const float*)d_in[2];
    const float*     b1   = (const float*)d_in[3];
    const float*     W2   = (const float*)d_in[4];
    const float*     b2   = (const float*)d_in[5];
    float* out = (float*)d_out;

    const int n_nodes = in_sizes[0] / HIDDEN;
    const int n_edges = in_sizes[1] / 2;

    unsigned short* zb  = (unsigned short*)d_ws;                 // n_nodes*128 bf16
    unsigned short* w1t = zb + (size_t)n_nodes * HIDDEN;         // 128*256 bf16

    const int nz4 = (n_nodes * HIDDEN) / 4;
    convert_z_kernel<<<(nz4 + 255) / 256, 256, 0, stream>>>(z, zb, nz4);
    convert_w1t_kernel<<<(2 * HIDDEN * HIDDEN) / 256, 256, 0, stream>>>(W1, w1t);

    const int blocks = (n_edges + 127) / 128;  // 128 edges per block (8 waves x 16)
    edge_decoder_kernel<<<blocks, 256, SW_BYTES, stream>>>(zb, edge, w1t, b1, W2,
                                                           b2, out, n_edges);
}
